// SPC32_15298673508954
// MI455X (gfx1250) — compile-verified, matrix operand order checked by reading, not run on hardware
//
#include <hip/hip_runtime.h>

// ---------------- problem constants ----------------
#define NB   16      // batch
#define CC   49      // channels (= M rows of both GEMMs)
#define HWD  81      // H*W
#define MS   24      // conv output channels
#define DD   1176    // MS*CC
#define KP   96      // padded HW (K of GEMM1 / N of GEMM2), 3 or 6 tiles of 16/32
#define DP   1184    // padded DD (N of GEMM1 / K of GEMM2), 74 tiles of 16, 37 of 32
#define MP   64      // padded CC
#define K7   7
#define PADC 3
#define EPSB 1e-5f
#define SLOPE 0.01f

typedef __attribute__((ext_vector_type(16))) __bf16 v16bf;
typedef __attribute__((ext_vector_type(8)))  float  v8f;

union Frag { v16bf v; uint4 q[2]; };

// ---------------------------------------------------------------------------
// Kernel 1: depth-conv over C (K=7, pad 3) + bias + softmax over 81 spatial
// positions. One wave per (n, d=m*C+c) row; lane covers hw = {l, l+32, l+64}.
// Output row stored bf16, zero-padded to KP=96. Optionally applies BatchNorm
// (stats + affine) to the *input* inline so BN1 never materializes a tensor.
// ---------------------------------------------------------------------------
__global__ __launch_bounds__(256)
void mask_softmax_kernel(const float* __restrict__ src,    // (NB,CC,HWD)
                         const float* __restrict__ w,      // (MS,K7)
                         const float* __restrict__ b,      // (MS)
                         const float* __restrict__ stats,  // (CC,2) mu,rsig
                         const float* __restrict__ gamma,  // (CC)
                         const float* __restrict__ beta,   // (CC)
                         int useBN,
                         __bf16* __restrict__ mask)        // (NB*DD, KP)
{
  const int wave = threadIdx.x >> 5;
  const int lane = threadIdx.x & 31;
  const int row  = blockIdx.x * 8 + wave;      // n*DD + dd
  if (row >= NB * DD) return;
  const int n  = row / DD;
  const int dd = row - n * DD;
  const int m  = dd / CC;
  const int c  = dd - m * CC;

  float v[3];
#pragma unroll
  for (int i = 0; i < 3; ++i) {
    const int hw = lane + 32 * i;
    float acc = b[m];
    if (hw < HWD) {
#pragma unroll
      for (int k = 0; k < K7; ++k) {
        const int cs = c + k - PADC;
        if (cs >= 0 && cs < CC) {
          float xv = src[((size_t)n * CC + cs) * HWD + hw];
          if (useBN)
            xv = (xv - stats[cs * 2]) * stats[cs * 2 + 1] * gamma[cs] + beta[cs];
          acc += w[m * K7 + k] * xv;
        }
      }
      v[i] = acc;
    } else {
      v[i] = -3.0e38f;
    }
  }
  float mx = fmaxf(fmaxf(v[0], v[1]), v[2]);
#pragma unroll
  for (int off = 16; off > 0; off >>= 1) mx = fmaxf(mx, __shfl_xor(mx, off));
  float e[3], s = 0.f;
#pragma unroll
  for (int i = 0; i < 3; ++i) {
    const int hw = lane + 32 * i;
    e[i] = (hw < HWD) ? __expf(v[i] - mx) : 0.f;
    s += e[i];
  }
#pragma unroll
  for (int off = 16; off > 0; off >>= 1) s += __shfl_xor(s, off);
  const float inv = 1.0f / s;

  __bf16* mrow = mask + (size_t)row * KP;
#pragma unroll
  for (int i = 0; i < 3; ++i) {
    const int hw = lane + 32 * i;           // covers 0..95 exactly
    const float val = (hw < HWD) ? e[i] * inv : 0.f;
    mrow[hw] = (__bf16)val;
  }
}

// ---------------------------------------------------------------------------
// Kernel 2 (GEMM1): G = leaky_relu( X(49x81) @ mask^T(81x1176) ), padded to
// 64 x 1184, stored bf16. One block = 4 waves; wave w owns N-tile
// blockIdx.x*4+w and accumulates the full 64-row M strip (4 WMMA tiles).
// A (x_n, f32) is staged once into LDS as bf16 64x96.
// ---------------------------------------------------------------------------
__global__ __launch_bounds__(128)
void gemm_fk_kernel(const float*  __restrict__ x,     // (NB,CC,HWD)
                    const __bf16* __restrict__ mask,  // (NB*DD,KP)
                    __bf16* __restrict__ G)           // (NB,MP,DP)
{
  __shared__ __bf16 As[MP * KP];                       // 12 KB
  const int n    = blockIdx.y;
  const int lane = threadIdx.x & 31;
  const int wave = threadIdx.x >> 5;

  const float* xn = x + (size_t)n * CC * HWD;
  for (int idx = threadIdx.x; idx < MP * KP; idx += 128) {
    const int r = idx / KP, k = idx - r * KP;
    const float fv = (r < CC && k < HWD) ? xn[r * HWD + k] : 0.f;
    As[idx] = (__bf16)fv;
  }
  __syncthreads();

  const int ntile = blockIdx.x * 4 + wave;             // 0..73 valid
  if (ntile >= DP / 16) return;
  const int nn = ntile * 16 + (lane & 15);             // global d column
  const int kh = lane >> 4;
  const bool bvalid = (nn < DD);
  const __bf16* mrow = mask + ((size_t)n * DD + (bvalid ? nn : 0)) * KP;

  v8f acc[4] = {};
#pragma unroll
  for (int ks = 0; ks < KP / 32; ++ks) {               // 3 k-steps
    Frag bf;
    if (bvalid) {
      bf.q[0] = *(const uint4*)(mrow + ks * 32 + kh * 8);
      bf.q[1] = *(const uint4*)(mrow + ks * 32 + 16 + kh * 8);
    } else {
      bf.q[0] = make_uint4(0u, 0u, 0u, 0u);
      bf.q[1] = make_uint4(0u, 0u, 0u, 0u);
    }
#pragma unroll
    for (int mt = 0; mt < 4; ++mt) {
      Frag af;
      const int rr = mt * 16 + (lane & 15);
      af.q[0] = *(const uint4*)&As[rr * KP + ks * 32 + kh * 8];
      af.q[1] = *(const uint4*)&As[rr * KP + ks * 32 + 16 + kh * 8];
      acc[mt] = __builtin_amdgcn_wmma_f32_16x16x32_bf16(
          false, af.v, false, bf.v, (short)0, acc[mt], false, false);
    }
  }

  __bf16* Gn = G + (size_t)n * MP * DP;
#pragma unroll
  for (int mt = 0; mt < 4; ++mt) {
#pragma unroll
    for (int r = 0; r < 8; ++r) {
      const int mr = mt * 16 + r + ((lane >> 4) << 3);
      float fv = acc[mt][r];
      fv = (fv > 0.f) ? fv : fv * SLOPE;               // leaky_relu(fk)
      Gn[(size_t)mr * DP + ntile * 16 + (lane & 15)] = (__bf16)fv;
    }
  }
}

// ---------------------------------------------------------------------------
// Kernel 3 (GEMM2): att = G(49x1176) @ mask(1176x81) (+ optional residual x).
// One block per batch, 6 waves; wave w owns hw-tile w (N padded 96 = 6 tiles).
// A-slice (64x32 of G) staged per k-step into LDS via the CDNA5 async
// global->LDS copy engine (ASYNCcnt path); B gathered per lane from global.
// ---------------------------------------------------------------------------
__global__ __launch_bounds__(192)
void gemm_att_kernel(const __bf16* __restrict__ G,     // (NB,MP,DP)
                     const __bf16* __restrict__ mask,  // (NB*DD,KP)
                     const float*  __restrict__ xres,  // (NB,CC,HWD)
                     int addRes,
                     float* __restrict__ att)          // (NB,CC,HWD)
{
  __shared__ __bf16 Gs[MP * 32];                       // 4 KB per k-step
  const int n    = blockIdx.x;
  const int lane = threadIdx.x & 31;
  const int wave = threadIdx.x >> 5;                   // 0..5 = n-tile
  const __bf16* Gn = G + (size_t)n * MP * DP;
  const __bf16* mn = mask + (size_t)n * DD * KP;
  const int kh = lane >> 4;
  const int nn = wave * 16 + (lane & 15);              // hw column 0..95

  v8f acc[4] = {};
  for (int ks = 0; ks < DP / 32; ++ks) {               // 37 k-steps
    __syncthreads();                                    // LDS reuse fence
#if defined(__gfx1250__)
    // Async copy: 64x32 bf16 slice = 256 x 16B chunks. Each chunk is
    // 16B-aligned (G row stride 1184*2 = 2368 = 148*16 bytes).
    typedef __attribute__((address_space(3))) __bf16 lds_bf16;
    for (int cidx = threadIdx.x; cidx < (MP * 32) / 8; cidx += 192) {
      const int r  = cidx >> 2;            // row 0..63
      const int ko = (cidx & 3) * 8;       // k offset 0,8,16,24
      const __bf16* gsrc = Gn + (size_t)r * DP + ks * 32 + ko;
      const unsigned ldsoff =
          (unsigned)(size_t)(lds_bf16*)&Gs[r * 32 + ko];
      asm volatile("global_load_async_to_lds_b128 %0, %1, off"
                   :: "v"(ldsoff), "v"((unsigned long long)(size_t)gsrc)
                   : "memory");
    }
    asm volatile("s_wait_asynccnt 0x0" ::: "memory");
#else
    for (int idx = threadIdx.x; idx < MP * 32; idx += 192) {
      const int r = idx >> 5, k = idx & 31;
      Gs[idx] = Gn[(size_t)r * DP + ks * 32 + k];
    }
#endif
    __syncthreads();

    // prefetch next B slab into L2 (global_prefetch_b8 path)
    if (ks + 1 < DP / 32) {
      const int kn = (ks + 1) * 32 + (kh << 3);
      if (kn < DD) __builtin_prefetch(&mn[(size_t)kn * KP + nn], 0, 1);
    }

    Frag bf;
#pragma unroll
    for (int j = 0; j < 16; ++j) {
      const int kl = ((j >> 3) << 4) + (kh << 3) + (j & 7);
      const int kd = ks * 32 + kl;
      bf.v[j] = (kd < DD) ? mn[(size_t)kd * KP + nn] : (__bf16)0.0f;
    }
#pragma unroll
    for (int mt = 0; mt < 4; ++mt) {
      Frag af;
      const int rr = mt * 16 + (lane & 15);
      af.q[0] = *(const uint4*)&Gs[rr * 32 + kh * 8];
      af.q[1] = *(const uint4*)&Gs[rr * 32 + 16 + kh * 8];
      acc[mt] = __builtin_amdgcn_wmma_f32_16x16x32_bf16(
          false, af.v, false, bf.v, (short)0, acc[mt], false, false);
    }
  }

#pragma unroll
  for (int mt = 0; mt < 4; ++mt) {
#pragma unroll
    for (int r = 0; r < 8; ++r) {
      const int m = mt * 16 + r + ((lane >> 4) << 3);
      if (m < CC && nn < HWD) {
        const size_t o = ((size_t)n * CC + m) * HWD + nn;
        float fv = acc[mt][r];
        if (addRes) fv += xres[o];
        att[o] = fv;
      }
    }
  }
}

// ---------------------------------------------------------------------------
// Kernel 4: training-mode BN batch statistics per channel (biased variance).
// ---------------------------------------------------------------------------
__global__ __launch_bounds__(256)
void bn_stats_kernel(const float* __restrict__ att,    // (NB,CC,HWD)
                     float* __restrict__ stats)        // (CC,2) mu, rsig
{
  const int c = blockIdx.x;
  __shared__ float ssum[256], ssq[256];
  float s = 0.f, q = 0.f;
  for (int idx = threadIdx.x; idx < NB * HWD; idx += 256) {
    const int nn = idx / HWD, hw = idx - nn * HWD;
    const float v = att[((size_t)nn * CC + c) * HWD + hw];
    s += v; q += v * v;
  }
  ssum[threadIdx.x] = s; ssq[threadIdx.x] = q;
  __syncthreads();
  for (int off = 128; off > 0; off >>= 1) {
    if (threadIdx.x < off) {
      ssum[threadIdx.x] += ssum[threadIdx.x + off];
      ssq[threadIdx.x]  += ssq[threadIdx.x + off];
    }
    __syncthreads();
  }
  if (threadIdx.x == 0) {
    const float cnt = (float)(NB * HWD);
    const float mu  = ssum[0] / cnt;
    const float var = ssq[0] / cnt - mu * mu;
    stats[c * 2 + 0] = mu;
    stats[c * 2 + 1] = rsqrtf(var + EPSB);
  }
}

// ---------------------------------------------------------------------------
// Kernel 5: final BN apply -> d_out
// ---------------------------------------------------------------------------
__global__ __launch_bounds__(256)
void bn_apply_kernel(const float* __restrict__ in,
                     const float* __restrict__ stats,
                     const float* __restrict__ gamma,
                     const float* __restrict__ beta,
                     float* __restrict__ out, int total)
{
  const int i = blockIdx.x * 256 + threadIdx.x;
  if (i >= total) return;
  const int c = (i / HWD) % CC;
  out[i] = (in[i] - stats[c * 2]) * stats[c * 2 + 1] * gamma[c] + beta[c];
}

// ---------------------------------------------------------------------------
extern "C" void kernel_launch(void* const* d_in, const int* in_sizes, int n_in,
                              void* d_out, int out_size, void* d_ws, size_t ws_size,
                              hipStream_t stream) {
  const float* x      = (const float*)d_in[0];
  const float* w0     = (const float*)d_in[1];
  const float* b0     = (const float*)d_in[2];
  const float* gamma1 = (const float*)d_in[3];
  const float* beta1  = (const float*)d_in[4];
  const float* w2     = (const float*)d_in[5];
  const float* b2     = (const float*)d_in[6];
  const float* gamma2 = (const float*)d_in[7];
  const float* beta2  = (const float*)d_in[8];
  float* out = (float*)d_out;

  // workspace carve-up (256B aligned)
  char*  ws  = (char*)d_ws;
  size_t off = 0;
  auto carve = [&](size_t bytes) {
    size_t o = off;
    off = (off + bytes + 255) & ~(size_t)255;
    return o;
  };
  __bf16* maskb  = (__bf16*)(ws + carve((size_t)NB * DD * KP * 2)); // ~3.45 MB
  __bf16* Gbuf   = (__bf16*)(ws + carve((size_t)NB * MP * DP * 2)); // ~2.31 MB
  float*  att0   = (float*) (ws + carve((size_t)NB * CC * HWD * 4));
  float*  outraw = (float*) (ws + carve((size_t)NB * CC * HWD * 4));
  float*  stats1 = (float*) (ws + carve((size_t)CC * 2 * 4));
  float*  stats2 = (float*) (ws + carve((size_t)CC * 2 * 4));
  (void)ws_size; (void)n_in; (void)in_sizes; (void)out_size;

  const dim3 maskGrid((NB * DD) / 8);
  const dim3 g1Grid((DP / 16 + 3) / 4, NB);   // 19 x 16
  const int  total = NB * CC * HWD;

  // ---- block 1: mask0 = softmax(conv(x)) ; att0 = attend(mask0, x) ----
  mask_softmax_kernel<<<maskGrid, 256, 0, stream>>>(
      x, w0, b0, stats1, gamma1, beta1, /*useBN=*/0, maskb);
  gemm_fk_kernel<<<g1Grid, 128, 0, stream>>>(x, maskb, Gbuf);
  gemm_att_kernel<<<NB, 192, 0, stream>>>(Gbuf, maskb, x, /*addRes=*/0, att0);
  bn_stats_kernel<<<CC, 256, 0, stream>>>(att0, stats1);

  // ---- block 2: mask2 = softmax(conv(BN1(att0))) ; out = attend(mask2,x)+x
  mask_softmax_kernel<<<maskGrid, 256, 0, stream>>>(
      att0, w2, b2, stats1, gamma1, beta1, /*useBN=*/1, maskb);
  gemm_fk_kernel<<<g1Grid, 128, 0, stream>>>(x, maskb, Gbuf);
  gemm_att_kernel<<<NB, 192, 0, stream>>>(Gbuf, maskb, x, /*addRes=*/1, outraw);

  // ---- final BN ----
  bn_stats_kernel<<<CC, 256, 0, stream>>>(outraw, stats2);
  bn_apply_kernel<<<(total + 255) / 256, 256, 0, stream>>>(
      outraw, stats2, gamma2, beta2, out, total);
}
